// MLAMoETransformerBlock_86466281603209
// MI455X (gfx1250) — matrix operand order repaired and checked by reading, not compiled, not run on hardware
//
#include <hip/hip_runtime.h>
#include <cstdint>
#include <cstddef>

// ---------------- types ----------------
typedef __bf16 bf16;
typedef __attribute__((ext_vector_type(8)))  __bf16 bf16x8;
typedef __attribute__((ext_vector_type(16))) __bf16 v16bf;
typedef __attribute__((ext_vector_type(8)))  float  v8f;

#define DEV static __device__ __forceinline__

// A-fragment (16x32 bf16): lane<16 -> row=lane, K0..7 & K16..23 ; lane>=16 -> row=lane-16, K8..15 & K24..31
DEV v16bf ld_a16(const bf16* p) {
  bf16x8 lo = *(const bf16x8*)(p);
  bf16x8 hi = *(const bf16x8*)(p + 16);
  return __builtin_shufflevector(lo, hi, 0,1,2,3,4,5,6,7,8,9,10,11,12,13,14,15);
}
// B-fragment (32x16 bf16, N-major storage Bt[n][k]): lane<16 -> col=lane, K0..7 & K16..23 ; lane>=16 -> K8..15 & K24..31
DEV v16bf ld_b16(const bf16* p) {
  bf16x8 lo = *(const bf16x8*)(p);
  bf16x8 hi = *(const bf16x8*)(p + 16);
  return __builtin_shufflevector(lo, hi, 0,1,2,3,4,5,6,7,8,9,10,11,12,13,14,15);
}
DEV v8f wmma_bf16(v16bf a, v16bf b, v8f c) {
  return __builtin_amdgcn_wmma_f32_16x16x32_bf16(false, a, false, b, (short)0, c, false, false);
}
DEV v8f v8f_zero() {
  v8f z;
  #pragma unroll
  for (int r = 0; r < 8; ++r) z[r] = 0.f;
  return z;
}

// CDNA5 async global->LDS copy (ASYNCcnt-tracked), 16B per lane
DEV void async_ld16(uint32_t lds, const void* g) {
  asm volatile("global_load_async_to_lds_b128 %0, %1, off"
               :: "v"(lds), "v"(g) : "memory");
}
DEV void wait_async4() { asm volatile("s_wait_asynccnt 0x4" ::: "memory"); }
DEV void wait_async0() { asm volatile("s_wait_asynccnt 0x0" ::: "memory"); }

// ---------------- model constants ----------------
static constexpr int S_  = 1024;
static constexpr int D_  = 1024;
static constexpr int L_  = 512;
static constexpr int H_  = 16;
static constexpr int HD_ = 32;
static constexpr int F_  = 2048;
static constexpr int E_  = 8;
static constexpr int T_  = 2048;   // B*S tokens
static constexpr int TOT_ROUTED = 2 * T_; // always 4096 (top-2, distinct experts)

// ============================================================
// Weight transpose + fp32 -> bf16 :  out[C][R] = (bf16) in[R][C]
// ============================================================
__global__ __launch_bounds__(256) void transpose_to_bf16(
    const float* __restrict__ in, bf16* __restrict__ out, int R, int C)
{
  size_t boff = (size_t)blockIdx.z * R * C;
  in  += boff;
  out += boff;
  __shared__ float tile[32][33];
  int tx = threadIdx.x & 31, ty = threadIdx.x >> 5;
  int rb = blockIdx.y * 32, cb = blockIdx.x * 32;
  #pragma unroll
  for (int i = 0; i < 4; ++i)
    tile[ty + i*8][tx] = in[(size_t)(rb + ty + i*8) * C + cb + tx];
  __syncthreads();
  #pragma unroll
  for (int i = 0; i < 4; ++i)
    out[(size_t)(cb + ty + i*8) * R + rb + tx] = (bf16)tile[tx][ty + i*8];
}

// ============================================================
// RMSNorm (one block per token), optional fused router top-2
// ============================================================
template<bool ROUTER>
__global__ __launch_bounds__(256) void rmsnorm_kernel(
    const float* __restrict__ x, const float* __restrict__ g, bf16* __restrict__ y,
    const float* __restrict__ rw, int* __restrict__ gate_e, float* __restrict__ gate_w)
{
  int t = blockIdx.x;
  int tid = threadIdx.x;
  const float* xr = x + (size_t)t * D_;
  float v[4];
  float ss = 0.f;
  #pragma unroll
  for (int i = 0; i < 4; ++i) { v[i] = xr[tid + i*256]; ss += v[i]*v[i]; }
  #pragma unroll
  for (int o = 16; o > 0; o >>= 1) ss += __shfl_xor(ss, o, 32);
  __shared__ float wsum[8];
  if ((tid & 31) == 0) wsum[tid >> 5] = ss;
  __syncthreads();
  float tot = 0.f;
  #pragma unroll
  for (int i = 0; i < 8; ++i) tot += wsum[i];
  float rstd = rsqrtf(tot * (1.f / (float)D_) + 1e-5f);
  float yv[4];
  #pragma unroll
  for (int i = 0; i < 4; ++i) {
    int d = tid + i*256;
    yv[i] = v[i] * g[d] * rstd;
    y[(size_t)t * D_ + d] = (bf16)yv[i];
  }
  if (ROUTER) {
    __shared__ float lsh[256 * E_];
    float lg[E_];
    #pragma unroll
    for (int e = 0; e < E_; ++e) lg[e] = 0.f;
    #pragma unroll
    for (int i = 0; i < 4; ++i) {
      int d = tid + i*256;
      #pragma unroll
      for (int e = 0; e < E_; ++e) lg[e] += yv[i] * rw[d * E_ + e];
    }
    #pragma unroll
    for (int e = 0; e < E_; ++e) lsh[tid * E_ + e] = lg[e];
    __syncthreads();
    for (int s2 = 128; s2 > 0; s2 >>= 1) {   // fixed-order tree: deterministic
      if (tid < s2) {
        #pragma unroll
        for (int e = 0; e < E_; ++e) lsh[tid * E_ + e] += lsh[(tid + s2) * E_ + e];
      }
      __syncthreads();
    }
    if (tid == 0) {
      float best = -1e30f; int be = 0;
      for (int e = 0; e < E_; ++e) if (lsh[e] > best) { best = lsh[e]; be = e; }
      float second = -1e30f; int se = 0;
      for (int e = 0; e < E_; ++e) if (e != be && lsh[e] > second) { second = lsh[e]; se = e; }
      float p0 = 1.f / (1.f + __expf(second - best));
      gate_e[t*2 + 0] = be;  gate_e[t*2 + 1] = se;
      gate_w[t*2 + 0] = p0;  gate_w[t*2 + 1] = 1.f - p0;
    }
  }
}

// ============================================================
// Generic tiled bf16 WMMA GEMM with async double-buffered LDS staging.
// C(MxN) = A(MxK) * Bt(NxK)^T ; block tile 128x128 ; K-step 32
// MODE 0 bf16 store | 1 f32 | 2 f32+addend | 3 MoE gather->compact f32
// MODE 4 MoE compact A -> gated scatter into y0/y1
// ============================================================
template<int MODE>
__global__ __launch_bounds__(256) void gemm_bf16_kernel(
    const bf16* __restrict__ A, const bf16* __restrict__ Bt,
    int M, int N, int Kd,
    float* __restrict__ outF, bf16* __restrict__ outB,
    const float* __restrict__ addend,
    const int* __restrict__ counts, const int* __restrict__ seg,
    const int* __restrict__ tokslot, const float* __restrict__ gates,
    float* __restrict__ y0, float* __restrict__ y1,
    const bf16* __restrict__ zpad)
{
  int e     = blockIdx.z;
  int rbase = blockIdx.y * 128;
  int cbase = blockIdx.x * 128;
  int cnt   = M;
  int segb  = 0;
  if (MODE >= 3) {
    cnt  = counts[e];
    segb = seg[e];
    if (rbase >= cnt) return;                  // uniform exit
    Bt += (size_t)e * N * Kd;
  }

  __shared__ bf16 AsD[2 * 128 * 32];
  __shared__ bf16 BsD[2 * 128 * 32];
  __shared__ int   rowTok[128];
  __shared__ float rowGate[128];

  int tid = threadIdx.x, lane = tid & 31, w = tid >> 5;
  int wm = w & 3, wn = w >> 2;
  int half = lane >> 4, r16 = lane & 15;

  if (MODE == 3 && tid < 128) {
    int i = rbase + tid;
    rowTok[tid] = (i < cnt) ? (tokslot[e * T_ + i] & 0xFFFF) : -1;
  }
  if (MODE == 4 && tid < 128) {
    int i = rbase + tid;
    rowTok[tid]  = (i < cnt) ? tokslot[e * T_ + i] : -1;
    rowGate[tid] = (i < cnt) ? gates[e * T_ + i]   : 0.f;
  }
  __syncthreads();

  uint32_t asB = (uint32_t)(uintptr_t)(&AsD[0]);
  uint32_t bsB = (uint32_t)(uintptr_t)(&BsD[0]);

  auto issue = [&](int buf, int k0) {
    #pragma unroll
    for (int i = 0; i < 2; ++i) {
      int c = tid + i*256;
      int r = c >> 2, ch = c & 3;
      const bf16* gp;
      if (MODE == 3) {
        int gr = rowTok[r];
        gp = (gr >= 0) ? (A + (size_t)gr * Kd + k0 + ch*8) : (zpad + ch*8);
      } else if (MODE == 4) {
        int i2 = rbase + r;
        gp = (i2 < cnt) ? (A + (size_t)(segb + i2) * Kd + k0 + ch*8) : (zpad + ch*8);
      } else {
        gp = A + (size_t)(rbase + r) * Kd + k0 + ch*8;
      }
      async_ld16(asB + (uint32_t)(buf*8192 + (r*32 + ch*8)*2), gp);
    }
    #pragma unroll
    for (int i = 0; i < 2; ++i) {
      int c = tid + i*256;
      int r = c >> 2, ch = c & 3;
      async_ld16(bsB + (uint32_t)(buf*8192 + (r*32 + ch*8)*2),
                 Bt + (size_t)(cbase + r) * Kd + k0 + ch*8);
    }
  };

  v8f acc[2][4];
  #pragma unroll
  for (int i = 0; i < 2; ++i)
    #pragma unroll
    for (int j = 0; j < 4; ++j) acc[i][j] = v8f_zero();

  issue(0, 0);
  int nk = Kd >> 5;
  for (int t2 = 0; t2 < nk; ++t2) {
    int buf = t2 & 1;
    if (t2 + 1 < nk) {
      issue(buf ^ 1, (t2 + 1) << 5);
      wait_async4();                 // the 4 older issues (current buf) are done
    } else {
      wait_async0();
    }
    __syncthreads();

    const bf16* Asb = AsD + buf * 4096;
    const bf16* Bsb = BsD + buf * 4096;
    v16bf af[2], bfr[4];
    #pragma unroll
    for (int i = 0; i < 2; ++i)
      af[i] = ld_a16(Asb + (wm*32 + i*16 + r16) * 32 + half*8);
    #pragma unroll
    for (int j = 0; j < 4; ++j)
      bfr[j] = ld_b16(Bsb + (wn*64 + j*16 + r16) * 32 + half*8);
    #pragma unroll
    for (int i = 0; i < 2; ++i)
      #pragma unroll
      for (int j = 0; j < 4; ++j)
        acc[i][j] = wmma_bf16(af[i], bfr[j], acc[i][j]);
    __syncthreads();
  }

  // ---- epilogue (C layout: lane = col r16, vgpr r -> row r + half*8) ----
  #pragma unroll
  for (int i = 0; i < 2; ++i) {
    #pragma unroll
    for (int j = 0; j < 4; ++j) {
      #pragma unroll
      for (int r = 0; r < 8; ++r) {
        int inrow = wm*32 + i*16 + r + half*8;
        int col   = cbase + wn*64 + j*16 + r16;
        int grow  = rbase + inrow;
        float v   = acc[i][j][r];
        if (MODE == 0) {
          outB[(size_t)grow * N + col] = (bf16)v;
        } else if (MODE == 1) {
          outF[(size_t)grow * N + col] = v;
        } else if (MODE == 2) {
          outF[(size_t)grow * N + col] = v + addend[(size_t)grow * N + col];
        } else if (MODE == 3) {
          if (grow < cnt) outF[(size_t)(segb + grow) * N + col] = v;
        } else { // MODE 4
          if (grow < cnt) {
            int ts   = rowTok[inrow];
            int tok  = ts & 0xFFFF;
            int slot = ts >> 16;
            float* dst = slot ? y1 : y0;
            dst[(size_t)tok * N + col] = rowGate[inrow] * v;   // unique write -> deterministic
          }
        }
      }
    }
  }
}

// ============================================================
// RoPE: qf [T][L] f32 interleaved pairs -> q_bf [bh][s][hd] bf16
// ============================================================
__global__ __launch_bounds__(256) void rope_kernel(
    const float* __restrict__ in, bf16* __restrict__ outp)
{
  int p = blockIdx.x * 256 + threadIdx.x;          // < T*H*16 = 524288 pairs
  int i = p & 15, h = (p >> 4) & 15, s = (p >> 8) & 1023, b = p >> 18;
  float inv = __expf(-logf(10000.f) * (float)i * (1.f/16.f));
  float ang = (float)s * inv;
  float c = cosf(ang), sn = sinf(ang);
  size_t src = ((size_t)(b*S_ + s)) * L_ + h*HD_ + 2*i;
  float t1 = in[src], t2 = in[src + 1];
  size_t dst = (((size_t)(b*H_ + h)) * S_ + s) * HD_ + 2*i;
  outp[dst]     = (bf16)(t1*c - t2*sn);
  outp[dst + 1] = (bf16)(t1*sn + t2*c);
}

// V: vf [T][L] f32  ->  vT [bh][hd][s] bf16 (N-major for the P@V WMMA)
__global__ __launch_bounds__(256) void v_transpose_kernel(
    const float* __restrict__ vf, bf16* __restrict__ vT)
{
  int idx = blockIdx.x * 256 + threadIdx.x;        // < T*L = 1048576
  int hd = idx & 31, h = (idx >> 5) & 15, s = (idx >> 9) & 1023, b = idx >> 19;
  vT[(((size_t)(b*H_ + h)) * HD_ + hd) * S_ + s] =
      (bf16)vf[((size_t)(b*S_ + s)) * L_ + h*HD_ + hd];
}

// ============================================================
// Flash attention, A-layout online softmax.
// grid (S/128, B*H); 8 waves x 16 q-rows; 32-key tiles.
// Scores go through per-wave LDS once; each lane then owns one q-row in
// exactly A-fragment chunk order -> row reductions need 1 shfl_xor(16),
// exp() results feed the P fragment directly.
// ============================================================
__global__ __launch_bounds__(256) void attn_kernel(
    const bf16* __restrict__ q, const bf16* __restrict__ k,
    const bf16* __restrict__ vT, bf16* __restrict__ o)
{
  int bh = blockIdx.y;
  int w = threadIdx.x >> 5, lane = threadIdx.x & 31;
  int half = lane >> 4, r16 = lane & 15;
  int qb = blockIdx.x * 128 + w * 16;

  __shared__ float sS[8][16 * 32];   // per-wave raw scores
  __shared__ float sRow[8][16];      // per-wave row-broadcast slot
  float* pws  = &sS[w][0];
  float* prow = &sRow[w][0];

  v16bf qa = ld_a16(q + ((size_t)bh * S_ + qb + r16) * HD_ + half*8);

  v8f oacc[2];
  oacc[0] = v8f_zero(); oacc[1] = v8f_zero();
  float mA = -1e30f, lA = 0.f;       // stats for row (qb + r16), replicated in both halves
  int qrow = qb + r16;

  const float scale = 0.17677669529663687f;        // 1/sqrt(HD)
  int ntiles = blockIdx.x * 4 + 4;                 // causal limit, uniform in WG

  for (int kt = 0; kt < ntiles; ++kt) {
    int key0 = kt * 32;
    v16bf kb0 = ld_b16(k + ((size_t)bh * S_ + key0 +      r16) * HD_ + half*8);
    v16bf kb1 = ld_b16(k + ((size_t)bh * S_ + key0 + 16 + r16) * HD_ + half*8);
    v8f s0 = wmma_bf16(qa, kb0, v8f_zero());
    v8f s1 = wmma_bf16(qa, kb1, v8f_zero());

    // C layout -> LDS (row r+half*8, col j*16+r16)
    #pragma unroll
    for (int r = 0; r < 8; ++r) {
      pws[(r + half*8)*32 +      r16] = s0[r];
      pws[(r + half*8)*32 + 16 + r16] = s1[r];
    }
    __syncthreads();

    // A layout: lane owns row r16, K chunks half*8..+7 and 16+half*8..+7
    float a[16];
    #pragma unroll
    for (int i = 0; i < 8; ++i) a[i]     = pws[r16*32 + half*8 + i];
    #pragma unroll
    for (int i = 0; i < 8; ++i) a[8 + i] = pws[r16*32 + 16 + half*8 + i];

    float tm = -1e30f;
    #pragma unroll
    for (int i = 0; i < 16; ++i) {
      int col = key0 + (i < 8 ? half*8 + i : 16 + half*8 + (i - 8));
      a[i] = (col <= qrow) ? a[i] * scale : -1e30f;
      tm = fmaxf(tm, a[i]);
    }
    tm = fmaxf(tm, __shfl_xor(tm, 16, 32));
    float nm = fmaxf(mA, tm);
    float sc = __expf(mA - nm);
    float ts = 0.f;
    v16bf pa;
    #pragma unroll
    for (int i = 0; i < 16; ++i) {
      float p = __expf(a[i] - nm);
      ts += p;
      pa[i] = (bf16)p;
    }
    ts += __shfl_xor(ts, 16, 32);
    lA = lA * sc + ts;
    mA = nm;

    if (half == 0) prow[r16] = sc;   // broadcast per-row rescale factor
    __syncthreads();
    #pragma unroll
    for (int r = 0; r < 8; ++r) {
      float f = prow[r + half*8];
      oacc[0][r] *= f;
      oacc[1][r] *= f;
    }

    v16bf vb0 = ld_b16(vT + ((size_t)bh * HD_ +      r16) * S_ + key0 + half*8);
    v16bf vb1 = ld_b16(vT + ((size_t)bh * HD_ + 16 + r16) * S_ + key0 + half*8);
    oacc[0] = wmma_bf16(pa, vb0, oacc[0]);
    oacc[1] = wmma_bf16(pa, vb1, oacc[1]);
  }

  __syncthreads();
  if (half == 0) prow[r16] = lA;
  __syncthreads();
  float lr[8];
  #pragma unroll
  for (int r = 0; r < 8; ++r) lr[r] = prow[r + half*8];

  int b = bh >> 4, h = bh & 15;
  #pragma unroll
  for (int j = 0; j < 2; ++j)
    #pragma unroll
    for (int r = 0; r < 8; ++r) {
      int row = qb + r + half*8;
      int col = h * HD_ + j*16 + r16;
      o[((size_t)b * S_ + row) * L_ + col] = (bf16)(oacc[j][r] / lr[r]);
    }
}

// ============================================================
// MoE bookkeeping
// ============================================================
__global__ void moe_assign(const int* __restrict__ gate_e, const float* __restrict__ gate_w,
                           int* __restrict__ counts, int* __restrict__ tokslot,
                           float* __restrict__ gates)
{
  int t = blockIdx.x * 256 + threadIdx.x;
  if (t >= T_) return;
  #pragma unroll
  for (int k2 = 0; k2 < 2; ++k2) {
    int e = gate_e[t*2 + k2];
    int pos = atomicAdd(&counts[e], 1);
    tokslot[e * T_ + pos] = t | (k2 << 16);
    gates[e * T_ + pos]   = gate_w[t*2 + k2];
  }
}

__global__ void moe_prefix(const int* __restrict__ counts, int* __restrict__ seg)
{
  if (threadIdx.x == 0 && blockIdx.x == 0) {
    int a = 0;
    for (int e = 0; e < E_; ++e) { seg[e] = a; a += counts[e]; }
    seg[E_] = a;
  }
}

__global__ __launch_bounds__(256) void silu_mul_kernel(
    const float* __restrict__ h1, const float* __restrict__ h3,
    bf16* __restrict__ hid, int total)
{
  int i = blockIdx.x * 256 + threadIdx.x;
  if (i >= total) return;
  float a = h1[i], b = h3[i];
  hid[i] = (bf16)((a / (1.f + __expf(-a))) * b);
}

__global__ __launch_bounds__(256) void final_add_kernel(
    const float* __restrict__ x1, const float* __restrict__ y0,
    const float* __restrict__ y1, float* __restrict__ out, int total)
{
  int i = blockIdx.x * 256 + threadIdx.x;
  if (i >= total) return;
  out[i] = x1[i] + (y0[i] + y1[i]);   // fixed association -> deterministic
}

// ============================================================
// Host launcher
// ============================================================
extern "C" void kernel_launch(void* const* d_in, const int* in_sizes, int n_in,
                              void* d_out, int out_size, void* d_ws, size_t ws_size,
                              hipStream_t stream)
{
  (void)in_sizes; (void)n_in; (void)out_size; (void)ws_size;
  const float* x        = (const float*)d_in[0];
  const float* g1       = (const float*)d_in[1];
  const float* g2       = (const float*)d_in[2];
  const float* w_down   = (const float*)d_in[3];
  const float* wq       = (const float*)d_in[4];
  const float* wk       = (const float*)d_in[5];
  const float* wv       = (const float*)d_in[6];
  const float* w_up     = (const float*)d_in[7];
  const float* router_w = (const float*)d_in[8];
  const float* w1       = (const float*)d_in[9];
  const float* w3       = (const float*)d_in[10];
  const float* w2       = (const float*)d_in[11];
  float* out = (float*)d_out;

  char* ws = (char*)d_ws;
  size_t off = 0;
  auto alloc = [&](size_t bytes) -> void* {
    void* p = ws + off;
    off += (bytes + 255) & ~(size_t)255;
    return p;
  };

  bf16* wdownT = (bf16*)alloc((size_t)L_ * D_ * 2);
  bf16* wqT    = (bf16*)alloc((size_t)L_ * L_ * 2);
  bf16* wkT    = (bf16*)alloc((size_t)L_ * L_ * 2);
  bf16* wvT    = (bf16*)alloc((size_t)L_ * L_ * 2);
  bf16* wupT   = (bf16*)alloc((size_t)D_ * L_ * 2);
  bf16* w1T    = (bf16*)alloc((size_t)E_ * F_ * D_ * 2);
  bf16* w3T    = (bf16*)alloc((size_t)E_ * F_ * D_ * 2);
  bf16* w2T    = (bf16*)alloc((size_t)E_ * D_ * F_ * 2);
  bf16* h_bf   = (bf16*)alloc((size_t)T_ * D_ * 2);
  bf16* c_bf   = (bf16*)alloc((size_t)T_ * L_ * 2);
  float* qf    = (float*)alloc((size_t)T_ * L_ * 4);
  float* kf    = (float*)alloc((size_t)T_ * L_ * 4);
  float* vf    = (float*)alloc((size_t)T_ * L_ * 4);
  bf16* q_bf   = (bf16*)alloc((size_t)T_ * L_ * 2);
  bf16* k_bf   = (bf16*)alloc((size_t)T_ * L_ * 2);
  bf16* vT_bf  = (bf16*)alloc((size_t)T_ * L_ * 2);
  bf16* o_bf   = (bf16*)alloc((size_t)T_ * L_ * 2);
  float* x1    = (float*)alloc((size_t)T_ * D_ * 4);
  bf16* tnorm  = (bf16*)alloc((size_t)T_ * D_ * 2);
  int*   gate_e  = (int*)alloc((size_t)T_ * 2 * 4);
  float* gate_w  = (float*)alloc((size_t)T_ * 2 * 4);
  int*   counts  = (int*)alloc(E_ * 4);
  int*   seg     = (int*)alloc((E_ + 1) * 4);
  int*   tokslot = (int*)alloc((size_t)E_ * T_ * 4);
  float* gatesL  = (float*)alloc((size_t)E_ * T_ * 4);
  float* h1buf   = (float*)alloc((size_t)TOT_ROUTED * F_ * 4);
  float* h3buf   = (float*)alloc((size_t)TOT_ROUTED * F_ * 4);
  bf16*  hid     = (bf16*)alloc((size_t)TOT_ROUTED * F_ * 2);
  float* y0      = (float*)alloc((size_t)T_ * D_ * 4);
  float* y1      = (float*)alloc((size_t)T_ * D_ * 4);
  bf16*  zpad    = (bf16*)alloc(256);

  dim3 blk(256);

  (void)hipMemsetAsync(counts, 0, E_ * sizeof(int), stream);
  (void)hipMemsetAsync(zpad, 0, 256, stream);

  // ---- weight prep (fp32 -> bf16, N-major) ----
  transpose_to_bf16<<<dim3(L_/32, D_/32, 1),  blk, 0, stream>>>(w_down, wdownT, D_, L_);
  transpose_to_bf16<<<dim3(L_/32, L_/32, 1),  blk, 0, stream>>>(wq, wqT, L_, L_);
  transpose_to_bf16<<<dim3(L_/32, L_/32, 1),  blk, 0, stream>>>(wk, wkT, L_, L_);
  transpose_to_bf16<<<dim3(L_/32, L_/32, 1),  blk, 0, stream>>>(wv, wvT, L_, L_);
  transpose_to_bf16<<<dim3(D_/32, L_/32, 1),  blk, 0, stream>>>(w_up, wupT, L_, D_);
  transpose_to_bf16<<<dim3(F_/32, D_/32, E_), blk, 0, stream>>>(w1, w1T, D_, F_);
  transpose_to_bf16<<<dim3(F_/32, D_/32, E_), blk, 0, stream>>>(w3, w3T, D_, F_);
  transpose_to_bf16<<<dim3(D_/32, F_/32, E_), blk, 0, stream>>>(w2, w2T, F_, D_);

  // ---- attention path ----
  rmsnorm_kernel<false><<<T_, blk, 0, stream>>>(x, g1, h_bf, nullptr, nullptr, nullptr);

  gemm_bf16_kernel<0><<<dim3(L_/128, T_/128, 1), blk, 0, stream>>>(
      h_bf, wdownT, T_, L_, D_, nullptr, c_bf, nullptr,
      nullptr, nullptr, nullptr, nullptr, nullptr, nullptr, zpad);

  gemm_bf16_kernel<1><<<dim3(L_/128, T_/128, 1), blk, 0, stream>>>(
      c_bf, wqT, T_, L_, L_, qf, nullptr, nullptr,
      nullptr, nullptr, nullptr, nullptr, nullptr, nullptr, zpad);
  gemm_bf16_kernel<1><<<dim3(L_/128, T_/128, 1), blk, 0, stream>>>(
      c_bf, wkT, T_, L_, L_, kf, nullptr, nullptr,
      nullptr, nullptr, nullptr, nullptr, nullptr, nullptr, zpad);
  gemm_bf16_kernel<1><<<dim3(L_/128, T_/128, 1), blk, 0, stream>>>(
      c_bf, wvT, T_, L_, L_, vf, nullptr, nullptr,
      nullptr, nullptr, nullptr, nullptr, nullptr, nullptr, zpad);

  rope_kernel<<<(T_*H_*16)/256, blk, 0, stream>>>(qf, q_bf);
  rope_kernel<<<(T_*H_*16)/256, blk, 0, stream>>>(kf, k_bf);
  v_transpose_kernel<<<(T_*L_)/256, blk, 0, stream>>>(vf, vT_bf);

  attn_kernel<<<dim3(S_/128, (T_/S_)*H_), blk, 0, stream>>>(q_bf, k_bf, vT_bf, o_bf);

  gemm_bf16_kernel<2><<<dim3(D_/128, T_/128, 1), blk, 0, stream>>>(
      o_bf, wupT, T_, D_, L_, x1, nullptr, x,
      nullptr, nullptr, nullptr, nullptr, nullptr, nullptr, zpad);

  // ---- MoE path ----
  rmsnorm_kernel<true><<<T_, blk, 0, stream>>>(x1, g2, tnorm, router_w, gate_e, gate_w);
  moe_assign<<<T_/256, blk, 0, stream>>>(gate_e, gate_w, counts, tokslot, gatesL);
  moe_prefix<<<1, 32, 0, stream>>>(counts, seg);

  gemm_bf16_kernel<3><<<dim3(F_/128, T_/128, E_), blk, 0, stream>>>(
      tnorm, w1T, T_, F_, D_, h1buf, nullptr, nullptr,
      counts, seg, tokslot, gatesL, nullptr, nullptr, zpad);
  gemm_bf16_kernel<3><<<dim3(F_/128, T_/128, E_), blk, 0, stream>>>(
      tnorm, w3T, T_, F_, D_, h3buf, nullptr, nullptr,
      counts, seg, tokslot, gatesL, nullptr, nullptr, zpad);

  silu_mul_kernel<<<(TOT_ROUTED * F_)/256, blk, 0, stream>>>(
      h1buf, h3buf, hid, TOT_ROUTED * F_);

  gemm_bf16_kernel<4><<<dim3(D_/128, T_/128, E_), blk, 0, stream>>>(
      hid, w2T, T_, D_, F_, nullptr, nullptr, nullptr,
      counts, seg, tokslot, gatesL, y0, y1, zpad);

  final_add_kernel<<<(T_*D_)/256, blk, 0, stream>>>(x1, y0, y1, out, T_ * D_);
}
